// SampleSelfAttention_6244882448561
// MI455X (gfx1250) — matrix operand order repaired and checked by reading, not compiled, not run on hardware
//
#include <hip/hip_runtime.h>
#include <hip/hip_bf16.h>
#include <math.h>

// Problem constants (match reference)
#define NS   4096   // num slots
#define ED   1024   // feat dim
#define E3   3072   // 3*E
#define HH   8      // heads
#define DH   128    // head dim
#define NEGV (-1e30f)

typedef __bf16 bf16_t;
typedef __bf16 v16bf __attribute__((ext_vector_type(16)));
typedef __bf16 v8bf  __attribute__((ext_vector_type(8)));
typedef float  v8f   __attribute__((ext_vector_type(8)));
typedef float  v4f   __attribute__((ext_vector_type(4)));
typedef unsigned int u32x4 __attribute__((ext_vector_type(4)));
typedef int    i32x4 __attribute__((ext_vector_type(4)));
typedef int    i32x8 __attribute__((ext_vector_type(8)));

__device__ __forceinline__ v16bf bf_combine(v8bf lo, v8bf hi) {
  return __builtin_shufflevector(lo, hi, 0,1,2,3,4,5,6,7,8,9,10,11,12,13,14,15);
}

// A-fragment: 16(M) x 32(K) bf16, source row-major [row, K], per ISA 7.12.2:
// lanes 0-15 row M=lane: v0-3 K=0-7, v4-7 K=16-23; lanes 16-31: K=8-15 / K=24-31.
__device__ __forceinline__ v16bf load_a_frag(const bf16_t* base, int ld, int lane) {
  int r = lane & 15, half = lane >> 4;
  const bf16_t* p = base + (size_t)r * ld + half * 8;
  v8bf lo = *(const v8bf*)p;          // K = half*8 + 0..7
  v8bf hi = *(const v8bf*)(p + 16);   // K = 16 + half*8 + 0..7
  return bf_combine(lo, hi);
}

// B-fragment: 32(K) x 16(N) bf16 where we hold B^T row-major [n, K]:
// lane n holds column n; lanes 0-15 K=0-7,16-23; lanes 16-31 K=8-15,24-31 (matches A).
__device__ __forceinline__ v16bf load_b_frag(const bf16_t* base, int ld, int lane) {
  int n = lane & 15, kh = lane >> 4;
  const bf16_t* p = base + (size_t)n * ld + kh * 8;
  v8bf lo = *(const v8bf*)p;
  v8bf hi = *(const v8bf*)(p + 16);
  return bf_combine(lo, hi);
}

// ---------------- fp32 -> bf16 convert (vectorized b128 in/out) ----------------
__global__ void cvt_bf16_kernel(const float* __restrict__ src, bf16_t* __restrict__ dst, int n) {
  int i = (blockIdx.x * blockDim.x + threadIdx.x) * 8;
  if (i >= n) return;
  v4f a = *(const v4f*)(src + i);
  v4f b = *(const v4f*)(src + i + 4);
  v8bf o;
#pragma unroll
  for (int j = 0; j < 4; ++j) { o[j] = (bf16_t)a[j]; o[4 + j] = (bf16_t)b[j]; }
  *(v8bf*)(dst + i) = o;
}

// ---------------- bf16 GEMM: out = A[M,K] @ B[NC,K]^T + bias ----------------
// Block = 8 waves; the 16-row A tile (16 x K bf16, K=1024 -> 32 KB) is staged
// into LDS once per block by the Tensor Data Mover, then all waves feed WMMA
// A-frags from LDS (ds_load_b128) while B weight fragments stream from L2.
// Ping-pong double buffering (unroll-by-2) avoids register rotation copies.
__global__ void __launch_bounds__(256)
gemm_bf16_kernel(const bf16_t* __restrict__ A, const bf16_t* __restrict__ B,
                 const float* __restrict__ bias,
                 bf16_t* __restrict__ outbf, float* __restrict__ outf,
                 const float* __restrict__ resid,
                 int M, int K, int NC) {
  __shared__ __align__(16) bf16_t Atile[16 * 1024];  // 16 rows x K (K==1024)

  int lane = threadIdx.x & 31;
  int wid  = threadIdx.x >> 5;
  int row0 = blockIdx.x * 16;
  int col0 = (blockIdx.y * (blockDim.x >> 5) + wid) * 64;

  // ---- TDM: stage A[row0:row0+16, 0:K] -> LDS (one issue per block) ----
  if (threadIdx.x == 0) {
    unsigned long long ga = (unsigned long long)(const void*)(A + (size_t)row0 * K);
    void* lp = (void*)&Atile[0];
    unsigned lds_off = (unsigned)(unsigned long long)lp;  // low 32b = LDS byte offset
    u32x4 g0;
    g0[0] = 1u;                                   // count=1 (valid descriptor)
    g0[1] = lds_off;                              // lds_addr
    g0[2] = (unsigned int)(ga & 0xFFFFFFFFull);   // global_addr[31:0]
    g0[3] = (unsigned int)((ga >> 32) & 0x1FFFFFFull) | 0x80000000u; // addr[56:32] | type=2
    i32x8 g1;
    g1[0] = 0x00010000;                           // data_size=1 (2 bytes/elt)
    g1[1] = (K & 0xFFFF) << 16;                   // tensor_dim0[15:0]
    g1[2] = ((K >> 16) & 0xFFFF) | (16 << 16);    // tensor_dim0[31:16] | tensor_dim1=16
    g1[3] = (K & 0xFFFF) << 16;                   // tile_dim0 = K
    g1[4] = 16;                                   // tile_dim1 = 16 rows
    g1[5] = K;                                    // tensor_dim0_stride = K
    g1[6] = 0;
    g1[7] = 0;
    i32x4 g2 = {0, 0, 0, 0};
    i32x4 g3 = {0, 0, 0, 0};
    // %4 (LDS pointer) escapes Atile into the asm so the "memory" clobber is
    // assumed to write it, keeping the LDS A-fragment loads alive.
    asm volatile("tensor_load_to_lds %0, %1, %2, %3"
                 : : "s"(g0), "s"(g1), "s"(g2), "s"(g3), "v"(lp) : "memory");
    __builtin_amdgcn_s_wait_tensorcnt(0);
  }
  __syncthreads();

  // ---- main loop: ping-pong buffers, unrolled by 2 (no rotation copies) ----
  const bf16_t* Bw = B + (size_t)col0 * K;
  v8f acc[4] = {};
  v16bf aA = load_a_frag(&Atile[0], 1024, lane);
  v16bf aB;
  v16bf bA[4], bB[4];
#pragma unroll
  for (int t = 0; t < 4; ++t) bA[t] = load_b_frag(Bw + (size_t)t * 16 * K, K, lane);

  for (int k0 = 0; k0 < K; k0 += 64) {
    // prefetch k0+32 (always in range: K % 64 == 0), compute with k0
    aB = load_a_frag(&Atile[k0 + 32], 1024, lane);
#pragma unroll
    for (int t = 0; t < 4; ++t)
      bB[t] = load_b_frag(Bw + (size_t)t * 16 * K + (k0 + 32), K, lane);
    __builtin_prefetch(Bw + (size_t)k0 + 128, 0, 1);  // global_prefetch
#pragma unroll
    for (int t = 0; t < 4; ++t)
      acc[t] = __builtin_amdgcn_wmma_f32_16x16x32_bf16(false, aA, false, bA[t],
                                                       (short)0, acc[t], false, false);
    // prefetch k0+64 (clip on last iter), compute with k0+32
    int kn = (k0 + 64 < K) ? k0 + 64 : 0;
    aA = load_a_frag(&Atile[kn], 1024, lane);
#pragma unroll
    for (int t = 0; t < 4; ++t)
      bA[t] = load_b_frag(Bw + (size_t)t * 16 * K + kn, K, lane);
#pragma unroll
    for (int t = 0; t < 4; ++t)
      acc[t] = __builtin_amdgcn_wmma_f32_16x16x32_bf16(false, aB, false, bB[t],
                                                       (short)0, acc[t], false, false);
  }

  int n = lane & 15, half = lane >> 4;
#pragma unroll
  for (int t = 0; t < 4; ++t) {
    int c = col0 + t * 16 + n;
    float bs = bias[c];
#pragma unroll
    for (int g = 0; g < 8; ++g) {
      int m = row0 + g + 8 * half;
      float v = acc[t][g] + bs;
      if (outbf) {
        outbf[(size_t)m * NC + c] = (bf16_t)v;
      } else {
        if (resid) v += resid[(size_t)m * NC + c];
        outf[(size_t)m * NC + c] = v;
      }
    }
  }
}

// ---------------- V transpose: qkv[n, 2E + e] -> vT[e, n]  (e = h*DH + d) ----------------
__global__ void vtrans_kernel(const bf16_t* __restrict__ qkv, bf16_t* __restrict__ vT) {
  int i = blockIdx.x * blockDim.x + threadIdx.x;
  if (i >= NS * ED) return;
  int n = i >> 10;          // slot
  int e = i & (ED - 1);     // h*DH + d
  vT[(size_t)e * NS + n] = qkv[(size_t)n * E3 + 2 * ED + e];
}

// ---------------- masked flash attention ----------------
// grid (NS/16, HH/4), block 128 (4 waves). Wave w handles head blockIdx.y*4+w,
// rows [16*blockIdx.x, +16). Sorted batch_idx -> binary-search live col range.
#define WPB 4
__global__ void attn_kernel(const bf16_t* __restrict__ qkv, const bf16_t* __restrict__ vT,
                            const int* __restrict__ bidx, bf16_t* __restrict__ ctx) {
  __shared__ __align__(16) float  Sbuf[WPB][16][32];
  __shared__ __align__(16) bf16_t Pbuf[WPB][16][32];
  __shared__ float mbuf[WPB][16], lbuf[WPB][16], abuf[WPB][16];
  __shared__ int   rbuf[WPB][16];

  int lane = threadIdx.x & 31;
  int w    = threadIdx.x >> 5;
  int h    = blockIdx.y * WPB + w;
  int row0 = blockIdx.x * 16;
  int n    = lane & 15, half = lane >> 4;
  const float scale = 0.08838834764831845f;  // 1/sqrt(128)

  if (lane < 16) {
    mbuf[w][lane] = -3.0e38f;
    lbuf[w][lane] = 0.0f;
    rbuf[w][lane] = bidx[row0 + lane];
  }
  // Preload Q fragments (16 x 128 bf16 -> 4 A-frags)
  v16bf qf[4];
#pragma unroll
  for (int kk = 0; kk < 4; ++kk)
    qf[kk] = load_a_frag(qkv + (size_t)row0 * E3 + h * DH + kk * 32, E3, lane);

  v8f o[8] = {};
  int rb[8];
#pragma unroll
  for (int g = 0; g < 8; ++g) rb[g] = rbuf[w][g + 8 * half];

  // Live column range: all cols whose batch in [b_lo, b_hi] (contiguous, sorted).
  int b_lo = rbuf[w][0];
  int b_hi = rbuf[w][15];
  int lo = 0, hi = NS;
  while (lo < hi) { int mid = (lo + hi) >> 1; if (bidx[mid] < b_lo) lo = mid + 1; else hi = mid; }
  int cstart = lo & ~31;
  lo = cstart; hi = NS;
  while (lo < hi) { int mid = (lo + hi) >> 1; if (bidx[mid] <= b_hi) lo = mid + 1; else hi = mid; }
  int cend = lo;

  for (int c0 = cstart; c0 < cend; c0 += 32) {
    // ---- scores: S(16x32) = Q(16x128) @ K^T ----
    v8f s0 = {}, s1 = {};
#pragma unroll
    for (int kk = 0; kk < 4; ++kk) {
      v16bf kf0 = load_b_frag(qkv + (size_t)c0 * E3 + ED + h * DH + kk * 32, E3, lane);
      s0 = __builtin_amdgcn_wmma_f32_16x16x32_bf16(false, qf[kk], false, kf0, (short)0, s0, false, false);
      v16bf kf1 = load_b_frag(qkv + (size_t)(c0 + 16) * E3 + ED + h * DH + kk * 32, E3, lane);
      s1 = __builtin_amdgcn_wmma_f32_16x16x32_bf16(false, qf[kk], false, kf1, (short)0, s1, false, false);
    }
    // ---- mask + stage to LDS ----
    int cb0 = bidx[c0 + n];
    int cb1 = bidx[c0 + 16 + n];
#pragma unroll
    for (int g = 0; g < 8; ++g) {
      int m = g + 8 * half;
      float v0 = s0[g] * scale; if (rb[g] != cb0) v0 = NEGV;
      float v1 = s1[g] * scale; if (rb[g] != cb1) v1 = NEGV;
      Sbuf[w][m][n]      = v0;
      Sbuf[w][m][16 + n] = v1;
    }
    // ---- per-row online softmax (lanes 0-15; same wave => DS in-order) ----
    if (lane < 16) {
      float mo = mbuf[w][lane];
      float tm = mo;
      float sv[32];
#pragma unroll
      for (int j = 0; j < 32; ++j) { sv[j] = Sbuf[w][lane][j]; tm = fmaxf(tm, sv[j]); }
      float alpha = __expf(mo - tm);
      float sum = 0.0f;
#pragma unroll
      for (int j = 0; j < 32; ++j) {
        float p = (sv[j] > -1e29f) ? __expf(sv[j] - tm) : 0.0f;
        sum += p;
        Pbuf[w][lane][j] = (bf16_t)p;
      }
      mbuf[w][lane] = tm;
      lbuf[w][lane] = lbuf[w][lane] * alpha + sum;
      abuf[w][lane] = alpha;
    }
    // ---- rescale accumulator ----
    float al[8];
#pragma unroll
    for (int g = 0; g < 8; ++g) al[g] = abuf[w][g + 8 * half];
#pragma unroll
    for (int dt = 0; dt < 8; ++dt)
#pragma unroll
      for (int g = 0; g < 8; ++g) o[dt][g] *= al[g];
    // ---- P A-frag from LDS; O += P(16x32) @ V(32x128) per 16-wide d tile ----
    {
      int r = lane & 15;
      v8bf plo = *(const v8bf*)&Pbuf[w][r][half * 8];
      v8bf phi = *(const v8bf*)&Pbuf[w][r][16 + half * 8];
      v16bf pf = bf_combine(plo, phi);
#pragma unroll
      for (int dt = 0; dt < 8; ++dt) {
        v16bf vf = load_b_frag(vT + ((size_t)h * DH + dt * 16) * NS + c0, NS, lane);
        o[dt] = __builtin_amdgcn_wmma_f32_16x16x32_bf16(false, pf, false, vf, (short)0, o[dt], false, false);
      }
    }
  }

  // ---- finalize: divide by l, write ctx bf16 [NS, ED] ----
  float li[8];
#pragma unroll
  for (int g = 0; g < 8; ++g) li[g] = 1.0f / lbuf[w][g + 8 * half];
#pragma unroll
  for (int dt = 0; dt < 8; ++dt)
#pragma unroll
    for (int g = 0; g < 8; ++g) {
      int m = row0 + g + 8 * half;
      ctx[(size_t)m * ED + h * DH + dt * 16 + n] = (bf16_t)(o[dt][g] * li[g]);
    }
}

// ---------------- LayerNorm over feat dim ----------------
__global__ void ln_kernel(const float* __restrict__ y0, const float* __restrict__ gm,
                          const float* __restrict__ bt, float* __restrict__ out) {
  __shared__ float rs[256], rq[256];
  int row = blockIdx.x, tid = threadIdx.x;
  const float* yr = y0 + (size_t)row * ED;
  float s = 0.f, q = 0.f;
  for (int j = tid; j < ED; j += 256) { float v = yr[j]; s += v; q += v * v; }
  rs[tid] = s; rq[tid] = q;
  __syncthreads();
  for (int off = 128; off > 0; off >>= 1) {
    if (tid < off) { rs[tid] += rs[tid + off]; rq[tid] += rq[tid + off]; }
    __syncthreads();
  }
  float mu = rs[0] * (1.0f / ED);
  float var = rq[0] * (1.0f / ED) - mu * mu;
  float rstd = rsqrtf(var + 1e-5f);
  float* orow = out + (size_t)row * ED;
  for (int j = tid; j < ED; j += 256) orow[j] = (yr[j] - mu) * rstd * gm[j] + bt[j];
}

extern "C" void kernel_launch(void* const* d_in, const int* in_sizes, int n_in,
                              void* d_out, int out_size, void* d_ws, size_t ws_size,
                              hipStream_t stream) {
  const float* slots  = (const float*)d_in[0];   // [1,N,E]
  const int*   bidx   = (const int*)d_in[1];     // [N]
  const float* seg    = (const float*)d_in[2];   // [N,32,32]
  const float* cur    = (const float*)d_in[3];   // [N,32,32]
  const float* w_in   = (const float*)d_in[4];   // [3E,E]
  const float* b_in   = (const float*)d_in[5];   // [3E]
  const float* w_out  = (const float*)d_in[6];   // [E,E]
  const float* b_out  = (const float*)d_in[7];   // [E]
  const float* gamma  = (const float*)d_in[8];
  const float* beta   = (const float*)d_in[9];
  float* out = (float*)d_out;

  char* ws = (char*)d_ws;
  bf16_t* xbf    = (bf16_t*)(ws);                       // N*E
  bf16_t* winbf  = (bf16_t*)(ws + 8388608);             // 3E*E
  bf16_t* woutbf = (bf16_t*)(ws + 14680064);            // E*E
  bf16_t* qkvbf  = (bf16_t*)(ws + 16777216);            // N*3E
  bf16_t* vTbf   = (bf16_t*)(ws + 41943040);            // E*N  ([H,D,N])
  bf16_t* ctxbf  = (bf16_t*)(ws + 50331648);            // N*E
  float*  y0     = (float*)(ws + 58720256);             // N*E fp32

  // 1) bf16 conversions (8 elements/thread, b128 traffic)
  cvt_bf16_kernel<<<(NS * ED / 8 + 255) / 256, 256, 0, stream>>>(slots, xbf, NS * ED);
  cvt_bf16_kernel<<<(E3 * ED / 8 + 255) / 256, 256, 0, stream>>>(w_in, winbf, E3 * ED);
  cvt_bf16_kernel<<<(ED * ED / 8 + 255) / 256, 256, 0, stream>>>(w_out, woutbf, ED * ED);

  // 2) QKV projection: qkv = x @ w_in^T + b_in  (bf16 out)
  gemm_bf16_kernel<<<dim3(NS / 16, E3 / 512), 256, 0, stream>>>(
      xbf, winbf, b_in, qkvbf, nullptr, nullptr, NS, ED, E3);

  // 3) transpose V for contiguous B-frags
  vtrans_kernel<<<(NS * ED + 255) / 256, 256, 0, stream>>>(qkvbf, vTbf);

  // 4) masked flash attention -> ctx bf16
  attn_kernel<<<dim3(NS / 16, HH / WPB), WPB * 32, 0, stream>>>(qkvbf, vTbf, bidx, ctxbf);

  // 5) out-proj + residual: y0 = x + ctx @ w_out^T + b_out  (fp32 out)
  gemm_bf16_kernel<<<dim3(NS / 16, ED / 512), 256, 0, stream>>>(
      ctxbf, woutbf, b_out, nullptr, y0, slots, NS, ED, ED);

  // 6) LayerNorm -> d_out[0 : N*E)
  ln_kernel<<<NS, 256, 0, stream>>>(y0, gamma, beta, out);

  // 7) pass-through outputs
  hipMemcpyAsync(out + (size_t)NS * ED, seg, (size_t)NS * 1024 * sizeof(float),
                 hipMemcpyDeviceToDevice, stream);
  hipMemcpyAsync(out + 2 * (size_t)NS * ED, cur, (size_t)NS * 1024 * sizeof(float),
                 hipMemcpyDeviceToDevice, stream);
}
